// EGNNLSPELayer_36335423324796
// MI455X (gfx1250) — compile-verified, matrix operand order checked by reading, not run on hardware
//
#include <hip/hip_runtime.h>
#include <hip/hip_bf16.h>

// ---------------- WMMA types & helpers (CDNA5 / gfx1250, wave32) ----------------
typedef __attribute__((ext_vector_type(16))) __bf16   v16bf;
typedef __attribute__((ext_vector_type(8)))  float    v8f;
typedef __attribute__((ext_vector_type(4)))  unsigned int uint4v;

union Frag { v16bf bf; uint4v u[2]; };

// float -> bf16 (round to nearest even)
__device__ __forceinline__ unsigned short f2bf(float f) {
    unsigned int u = __float_as_uint(f);
    u += 0x7FFFu + ((u >> 16) & 1u);
    return (unsigned short)(u >> 16);
}

__device__ __forceinline__ unsigned int f2bf2(float lo, float hi) {
    return (unsigned int)f2bf(lo) | ((unsigned int)f2bf(hi) << 16);
}

__device__ __forceinline__ float silu_f(float v) {
    return v / (1.0f + __expf(-v));   // one v_exp_f32
}

__device__ __forceinline__ float tanh_f(float v) {
#if __has_builtin(__builtin_amdgcn_tanhf)
    return __builtin_amdgcn_tanhf(v);             // hardware v_tanh_f32
#elif __has_builtin(__builtin_amdgcn_tanh_f32)
    return __builtin_amdgcn_tanh_f32(v);
#else
    float t = __expf(-2.0f * fabsf(v));           // stable fallback, one v_exp_f32
    float r = (1.0f - t) / (1.0f + t);
    return copysignf(r, v);
#endif
}

// A-fragment: lane L holds row m=L%16; slots 0..7 -> k = base+(hi?8:0)..,
// slots 8..15 -> k = base+(hi?24:16)..  (ISA 7.12.2, 16-bit A 16x32)
__device__ __forceinline__ v16bf load_a(const unsigned short* rowp, int kbase, int hi) {
    Frag f;
    f.u[0] = *(const uint4v*)(rowp + kbase + (hi ? 8 : 0));
    f.u[1] = *(const uint4v*)(rowp + kbase + (hi ? 24 : 16));
    return f.bf;
}

// B-fragment: pre-swizzled in global memory; 16 contiguous bf16 per lane.
__device__ __forceinline__ v16bf load_b(const unsigned short* __restrict__ wf,
                                        int chunk, int t, int lane) {
    const unsigned short* p = wf + ((size_t)(chunk * 4 + t) * 32 + lane) * 16;
    Frag f;
    f.u[0] = *(const uint4v*)(p);
    f.u[1] = *(const uint4v*)(p + 8);
    return f.bf;
}

__device__ __forceinline__ v8f wmma_bf16(v16bf a, v16bf b, v8f c) {
    return __builtin_amdgcn_wmma_f32_16x16x32_bf16(false, a, false, b, (short)0, c, false, false);
}

// ---------------- prep kernels ----------------
__global__ void zero_kernel(float* __restrict__ p, int n) {
    int i = blockIdx.x * blockDim.x + threadIdx.x;
    if (i < n) p[i] = 0.0f;
}

// Chunk map (each chunk = K-slab of 32, all 4 n-tiles, 32 lanes x 16 slots):
//  [0..8]  msg_w1 (K=257)   [9..10]  msg_w2 (64)
//  [11..15] mpos_w1 (129)   [16..17] mpos_w2 (64)
//  [18..23] upd_w1 (192)    [24..25] upd_w2 (64)
//  [26..29] upe_w1 (128)    [30..31] upe_w2 (64)
__global__ void weight_swizzle_kernel(
    const float* __restrict__ msg_w1, const float* __restrict__ msg_w2,
    const float* __restrict__ mpos_w1, const float* __restrict__ mpos_w2,
    const float* __restrict__ upd_w1, const float* __restrict__ upd_w2,
    const float* __restrict__ upe_w1, const float* __restrict__ upe_w2,
    unsigned short* __restrict__ wfrag)
{
    int tidx = blockIdx.x * blockDim.x + threadIdx.x;   // 0 .. 32*2048-1
    int chunk = tidx >> 11;
    int r     = tidx & 2047;
    int ntile = r >> 9;
    int lane  = (r >> 4) & 31;
    int s     = r & 15;

    const float* W; int Kact; int lc;
    if      (chunk < 9)  { W = msg_w1;  Kact = 257; lc = chunk;      }
    else if (chunk < 11) { W = msg_w2;  Kact = 64;  lc = chunk - 9;  }
    else if (chunk < 16) { W = mpos_w1; Kact = 129; lc = chunk - 11; }
    else if (chunk < 18) { W = mpos_w2; Kact = 64;  lc = chunk - 16; }
    else if (chunk < 24) { W = upd_w1;  Kact = 192; lc = chunk - 18; }
    else if (chunk < 26) { W = upd_w2;  Kact = 64;  lc = chunk - 24; }
    else if (chunk < 30) { W = upe_w1;  Kact = 128; lc = chunk - 26; }
    else                 { W = upe_w2;  Kact = 64;  lc = chunk - 30; }

    int k = lc * 32 + (s & 7) + ((s & 8) ? 16 : 0) + ((lane & 16) ? 8 : 0);
    int n = ntile * 16 + (lane & 15);
    float v = (k < Kact) ? W[(size_t)k * 64 + n] : 0.0f;
    wfrag[((size_t)(chunk * 4 + ntile) * 32 + lane) * 16 + s] = f2bf(v);
}

// pack a float4 into 4 bf16 (two dwords)
__device__ __forceinline__ void store_bf4(unsigned short* dst, float4 v) {
    uint2 p;
    p.x = f2bf2(v.x, v.y);
    p.y = f2bf2(v.z, v.w);
    *(uint2*)dst = p;   // ds_store_b64
}

// ---------------- edge kernel: 64 edges/block, message MLPs + scatter-add ----------------
__global__ __launch_bounds__(256) void edge_kernel(
    const float* __restrict__ x, const float* __restrict__ pos, const float* __restrict__ pe,
    const long long* __restrict__ ei,
    const float* __restrict__ msg_b1, const float* __restrict__ msg_b2,
    const float* __restrict__ mpos_b1, const float* __restrict__ mpos_b2,
    const unsigned short* __restrict__ wfrag,
    float* __restrict__ aggr, float* __restrict__ aggr_pos, int Etot)
{
    __shared__ alignas(16) unsigned short sMsg[4][16][288];  // [x_s|pe_s|x_r|pe_r|dist|pad]
    __shared__ alignas(16) unsigned short sPos[4][16][160];  // [pe_s|pe_r|dist|pad]
    __shared__ alignas(16) unsigned short sH[4][16][64];     // hidden bounce
    __shared__ int sRec[64];

    const int tid = threadIdx.x;

    // ---- stage edge state: 4 threads per edge, 16 cols each ----
    {
        int eloc = tid >> 2;          // 0..63
        int sub  = tid & 3;
        long long e  = (long long)blockIdx.x * 64 + eloc;
        long long ec = (e < Etot) ? e : (long long)Etot - 1;
        long long snd = ei[ec];
        long long rcv = ei[(long long)Etot + ec];
        int g = eloc >> 4, m = eloc & 15;
        if (sub == 0) {
            float dx = pos[snd * 3 + 0] - pos[rcv * 3 + 0];
            float dy = pos[snd * 3 + 1] - pos[rcv * 3 + 1];
            float dz = pos[snd * 3 + 2] - pos[rcv * 3 + 2];
            float dist = sqrtf(dx * dx + dy * dy + dz * dz);
            sRec[eloc] = (int)rcv;
            sMsg[g][m][256] = f2bf(dist);
            sPos[g][m][128] = f2bf(dist);
        }
        const float4* xs = (const float4*)(x + snd * 64);
        const float4* ps = (const float4*)(pe + snd * 64);
        const float4* xr = (const float4*)(x + rcv * 64);
        const float4* pr = (const float4*)(pe + rcv * 64);
        #pragma unroll
        for (int j = 0; j < 4; ++j) {
            int col = sub * 16 + j * 4;
            float4 a = xs[sub * 4 + j];
            float4 b = ps[sub * 4 + j];
            float4 c = xr[sub * 4 + j];
            float4 d = pr[sub * 4 + j];
            store_bf4(&sMsg[g][m][col],       a);
            store_bf4(&sMsg[g][m][64 + col],  b);
            store_bf4(&sMsg[g][m][128 + col], c);
            store_bf4(&sMsg[g][m][192 + col], d);
            store_bf4(&sPos[g][m][col],       b);
            store_bf4(&sPos[g][m][64 + col],  d);
        }
        for (int c2 = 257 + sub; c2 < 288; c2 += 4) sMsg[g][m][c2] = 0;
        for (int c2 = 129 + sub; c2 < 160; c2 += 4) sPos[g][m][c2] = 0;
    }
    __syncthreads();

    const int wave = tid >> 5;
    const int lane = tid & 31;
    const int t    = wave & 3;          // n-tile 0..3
    const int g0   = (wave >> 2) << 1;  // this wave's two edge groups
    const int g1   = g0 + 1;
    const int row  = lane & 15;
    const int hi   = lane >> 4;
    const int ng   = t * 16 + row;      // output column owned by this lane

    // ---- message MLP layer 1: [16x288] @ [288x64], B reused for 2 M-tiles ----
    v8f m1a = {}, m1b = {};
    #pragma unroll
    for (int c = 0; c < 9; ++c) {
        v16bf b  = load_b(wfrag, c, t, lane);
        v16bf a0 = load_a(&sMsg[g0][row][0], c * 32, hi);
        v16bf a1 = load_a(&sMsg[g1][row][0], c * 32, hi);
        m1a = wmma_bf16(a0, b, m1a);
        m1b = wmma_bf16(a1, b, m1b);
    }
    {
        float bias = msg_b1[ng];
        #pragma unroll
        for (int r2 = 0; r2 < 8; ++r2) {
            sH[g0][r2 + hi * 8][ng] = f2bf(silu_f(m1a[r2] + bias));
            sH[g1][r2 + hi * 8][ng] = f2bf(silu_f(m1b[r2] + bias));
        }
    }
    __syncthreads();

    // ---- message MLP layer 2 + silu + scatter-add ----
    v8f m2a = {}, m2b = {};
    #pragma unroll
    for (int c = 0; c < 2; ++c) {
        v16bf b  = load_b(wfrag, 9 + c, t, lane);
        v16bf a0 = load_a(&sH[g0][row][0], c * 32, hi);
        v16bf a1 = load_a(&sH[g1][row][0], c * 32, hi);
        m2a = wmma_bf16(a0, b, m2a);
        m2b = wmma_bf16(a1, b, m2b);
    }
    {
        float bias = msg_b2[ng];
        #pragma unroll
        for (int r2 = 0; r2 < 8; ++r2) {
            int m = r2 + hi * 8;
            long long e0 = (long long)blockIdx.x * 64 + g0 * 16 + m;
            if (e0 < Etot)
                atomicAdd(&aggr[(size_t)sRec[g0 * 16 + m] * 64 + ng], silu_f(m2a[r2] + bias));
            if (e0 + 16 < Etot)
                atomicAdd(&aggr[(size_t)sRec[g1 * 16 + m] * 64 + ng], silu_f(m2b[r2] + bias));
        }
    }
    __syncthreads();

    // ---- positional MLP layer 1: [16x160] @ [160x64] ----
    v8f p1a = {}, p1b = {};
    #pragma unroll
    for (int c = 0; c < 5; ++c) {
        v16bf b  = load_b(wfrag, 11 + c, t, lane);
        v16bf a0 = load_a(&sPos[g0][row][0], c * 32, hi);
        v16bf a1 = load_a(&sPos[g1][row][0], c * 32, hi);
        p1a = wmma_bf16(a0, b, p1a);
        p1b = wmma_bf16(a1, b, p1b);
    }
    {
        float bias = mpos_b1[ng];
        #pragma unroll
        for (int r2 = 0; r2 < 8; ++r2) {
            sH[g0][r2 + hi * 8][ng] = f2bf(tanh_f(p1a[r2] + bias));
            sH[g1][r2 + hi * 8][ng] = f2bf(tanh_f(p1b[r2] + bias));
        }
    }
    __syncthreads();

    // ---- positional MLP layer 2 + tanh + scatter-add ----
    v8f p2a = {}, p2b = {};
    #pragma unroll
    for (int c = 0; c < 2; ++c) {
        v16bf b  = load_b(wfrag, 16 + c, t, lane);
        v16bf a0 = load_a(&sH[g0][row][0], c * 32, hi);
        v16bf a1 = load_a(&sH[g1][row][0], c * 32, hi);
        p2a = wmma_bf16(a0, b, p2a);
        p2b = wmma_bf16(a1, b, p2b);
    }
    {
        float bias = mpos_b2[ng];
        #pragma unroll
        for (int r2 = 0; r2 < 8; ++r2) {
            int m = r2 + hi * 8;
            long long e0 = (long long)blockIdx.x * 64 + g0 * 16 + m;
            if (e0 < Etot)
                atomicAdd(&aggr_pos[(size_t)sRec[g0 * 16 + m] * 64 + ng], tanh_f(p2a[r2] + bias));
            if (e0 + 16 < Etot)
                atomicAdd(&aggr_pos[(size_t)sRec[g1 * 16 + m] * 64 + ng], tanh_f(p2b[r2] + bias));
        }
    }
}

// ---------------- node kernel: 64 nodes/block, update + update_pe MLPs ----------------
__global__ __launch_bounds__(256) void node_kernel(
    const float* __restrict__ x, const float* __restrict__ pe,
    const float* __restrict__ aggr, const float* __restrict__ aggr_pos,
    const float* __restrict__ upd_b1, const float* __restrict__ upd_b2,
    const float* __restrict__ upe_b1, const float* __restrict__ upe_b2,
    const unsigned short* __restrict__ wfrag,
    float* __restrict__ out, int Ntot)
{
    __shared__ alignas(16) unsigned short sUpd[4][16][192];  // [x|pe|aggr]
    __shared__ alignas(16) unsigned short sUpe[4][16][128];  // [pe|aggr_pos]
    __shared__ alignas(16) unsigned short sH[4][16][64];

    const int tid = threadIdx.x;
    {
        int nloc = tid >> 2;          // 0..63
        int sub  = tid & 3;
        int node = blockIdx.x * 64 + nloc;
        int nc   = (node < Ntot) ? node : (Ntot - 1);
        int g = nloc >> 4, m = nloc & 15;
        const float4* xp = (const float4*)(x + (size_t)nc * 64);
        const float4* pp = (const float4*)(pe + (size_t)nc * 64);
        const float4* ap = (const float4*)(aggr + (size_t)nc * 64);
        const float4* qp = (const float4*)(aggr_pos + (size_t)nc * 64);
        #pragma unroll
        for (int j = 0; j < 4; ++j) {
            int col = sub * 16 + j * 4;
            float4 a = xp[sub * 4 + j];
            float4 b = pp[sub * 4 + j];
            float4 c = ap[sub * 4 + j];
            float4 d = qp[sub * 4 + j];
            store_bf4(&sUpd[g][m][col],       a);
            store_bf4(&sUpd[g][m][64 + col],  b);
            store_bf4(&sUpd[g][m][128 + col], c);
            store_bf4(&sUpe[g][m][col],       b);
            store_bf4(&sUpe[g][m][64 + col],  d);
        }
    }
    __syncthreads();

    const int wave = tid >> 5;
    const int lane = tid & 31;
    const int t    = wave & 3;
    const int g0   = (wave >> 2) << 1;
    const int g1   = g0 + 1;
    const int row  = lane & 15;
    const int hi   = lane >> 4;
    const int ng   = t * 16 + row;

    // ---- update MLP layer 1: [16x192] @ [192x64], silu ----
    v8f u1a = {}, u1b = {};
    #pragma unroll
    for (int c = 0; c < 6; ++c) {
        v16bf b  = load_b(wfrag, 18 + c, t, lane);
        v16bf a0 = load_a(&sUpd[g0][row][0], c * 32, hi);
        v16bf a1 = load_a(&sUpd[g1][row][0], c * 32, hi);
        u1a = wmma_bf16(a0, b, u1a);
        u1b = wmma_bf16(a1, b, u1b);
    }
    {
        float bias = upd_b1[ng];
        #pragma unroll
        for (int r2 = 0; r2 < 8; ++r2) {
            sH[g0][r2 + hi * 8][ng] = f2bf(silu_f(u1a[r2] + bias));
            sH[g1][r2 + hi * 8][ng] = f2bf(silu_f(u1b[r2] + bias));
        }
    }
    __syncthreads();

    // ---- update MLP layer 2 (no activation) -> out[0 : N*64] ----
    v8f u2a = {}, u2b = {};
    #pragma unroll
    for (int c = 0; c < 2; ++c) {
        v16bf b  = load_b(wfrag, 24 + c, t, lane);
        v16bf a0 = load_a(&sH[g0][row][0], c * 32, hi);
        v16bf a1 = load_a(&sH[g1][row][0], c * 32, hi);
        u2a = wmma_bf16(a0, b, u2a);
        u2b = wmma_bf16(a1, b, u2b);
    }
    {
        float bias = upd_b2[ng];
        #pragma unroll
        for (int r2 = 0; r2 < 8; ++r2) {
            int m = r2 + hi * 8;
            int n0 = blockIdx.x * 64 + g0 * 16 + m;
            if (n0 < Ntot)      out[(size_t)n0 * 64 + ng]        = u2a[r2] + bias;
            if (n0 + 16 < Ntot) out[(size_t)(n0 + 16) * 64 + ng] = u2b[r2] + bias;
        }
    }
    __syncthreads();

    // ---- pe-update MLP layer 1: [16x128] @ [128x64], tanh ----
    v8f e1a = {}, e1b = {};
    #pragma unroll
    for (int c = 0; c < 4; ++c) {
        v16bf b  = load_b(wfrag, 26 + c, t, lane);
        v16bf a0 = load_a(&sUpe[g0][row][0], c * 32, hi);
        v16bf a1 = load_a(&sUpe[g1][row][0], c * 32, hi);
        e1a = wmma_bf16(a0, b, e1a);
        e1b = wmma_bf16(a1, b, e1b);
    }
    {
        float bias = upe_b1[ng];
        #pragma unroll
        for (int r2 = 0; r2 < 8; ++r2) {
            sH[g0][r2 + hi * 8][ng] = f2bf(tanh_f(e1a[r2] + bias));
            sH[g1][r2 + hi * 8][ng] = f2bf(tanh_f(e1b[r2] + bias));
        }
    }
    __syncthreads();

    // ---- pe-update MLP layer 2 + tanh -> out[N*64 : 2*N*64] ----
    v8f e2a = {}, e2b = {};
    #pragma unroll
    for (int c = 0; c < 2; ++c) {
        v16bf b  = load_b(wfrag, 30 + c, t, lane);
        v16bf a0 = load_a(&sH[g0][row][0], c * 32, hi);
        v16bf a1 = load_a(&sH[g1][row][0], c * 32, hi);
        e2a = wmma_bf16(a0, b, e2a);
        e2b = wmma_bf16(a1, b, e2b);
    }
    {
        float bias = upe_b2[ng];
        #pragma unroll
        for (int r2 = 0; r2 < 8; ++r2) {
            int m = r2 + hi * 8;
            int n0 = blockIdx.x * 64 + g0 * 16 + m;
            if (n0 < Ntot)
                out[(size_t)Ntot * 64 + (size_t)n0 * 64 + ng] = tanh_f(e2a[r2] + bias);
            if (n0 + 16 < Ntot)
                out[(size_t)Ntot * 64 + (size_t)(n0 + 16) * 64 + ng] = tanh_f(e2b[r2] + bias);
        }
    }
}

// ---------------- host launcher ----------------
extern "C" void kernel_launch(void* const* d_in, const int* in_sizes, int n_in,
                              void* d_out, int out_size, void* d_ws, size_t ws_size,
                              hipStream_t stream) {
    const float*     x       = (const float*)d_in[0];
    const float*     pos     = (const float*)d_in[1];
    const float*     pe      = (const float*)d_in[2];
    const long long* ei      = (const long long*)d_in[3];
    const float*     msg_w1  = (const float*)d_in[4];
    const float*     msg_b1  = (const float*)d_in[5];
    const float*     msg_w2  = (const float*)d_in[6];
    const float*     msg_b2  = (const float*)d_in[7];
    const float*     mpos_w1 = (const float*)d_in[8];
    const float*     mpos_b1 = (const float*)d_in[9];
    const float*     mpos_w2 = (const float*)d_in[10];
    const float*     mpos_b2 = (const float*)d_in[11];
    const float*     upd_w1  = (const float*)d_in[12];
    const float*     upd_b1  = (const float*)d_in[13];
    const float*     upd_w2  = (const float*)d_in[14];
    const float*     upd_b2  = (const float*)d_in[15];
    const float*     upe_w1  = (const float*)d_in[16];
    const float*     upe_b1  = (const float*)d_in[17];
    const float*     upe_w2  = (const float*)d_in[18];
    const float*     upe_b2  = (const float*)d_in[19];

    const int Ntot = in_sizes[0] / 64;   // 50000
    const int Etot = in_sizes[3] / 2;    // 800000

    float*          aggr     = (float*)d_ws;
    float*          aggr_pos = aggr + (size_t)Ntot * 64;
    unsigned short* wfrag    = (unsigned short*)(aggr_pos + (size_t)Ntot * 64);

    {
        int n = 2 * Ntot * 64;
        zero_kernel<<<(n + 255) / 256, 256, 0, stream>>>(aggr, n);
    }
    weight_swizzle_kernel<<<(32 * 2048) / 256, 256, 0, stream>>>(
        msg_w1, msg_w2, mpos_w1, mpos_w2, upd_w1, upd_w2, upe_w1, upe_w2, wfrag);

    edge_kernel<<<(Etot + 63) / 64, 256, 0, stream>>>(
        x, pos, pe, ei, msg_b1, msg_b2, mpos_b1, mpos_b2, wfrag, aggr, aggr_pos, Etot);

    node_kernel<<<(Ntot + 63) / 64, 256, 0, stream>>>(
        x, pe, aggr, aggr_pos, upd_b1, upd_b2, upe_b1, upe_b2, wfrag, (float*)d_out, Ntot);
}
